// Encoder_21603685499468
// MI455X (gfx1250) — matrix operand order repaired and checked by reading, not compiled, not run on hardware
//
#include <hip/hip_runtime.h>
#include <hip/hip_bf16.h>

typedef __attribute__((ext_vector_type(16))) _Float16 v16h;
typedef __attribute__((ext_vector_type(8)))  _Float16 v8h;
typedef __attribute__((ext_vector_type(8)))  float    v8f;

#define IN_DIM 512
#define BATCH 8
#define SEQ 2048
#define NROWS (BATCH * SEQ)          // 16384
#define QKV_N (3 * IN_DIM)           // 1536
#define LEAKY 0.01f
#define LN_EPS 1e-5f

// ---------------------------------------------------------------------------
// Fragment loaders matching the CDNA5 wave32 WMMA VGPR layouts (ISA 7.12.2).
// A (16x32 f16): lane L holds row M=L&15; K = hs*8+j (j=0..7) and 16+hs*8+j.
// B (32x16 f16): lane L holds col N=L&15; K = hs*8+j (j=0..7) and 16+hs*8+j. B operand is
// stored "transposed" ([N][K] row-major) so these are two 8-f16 runs.
// C/D (16x16 f32): lane L, vgpr r -> (M = r + 8*hs, N = L&15).
// ---------------------------------------------------------------------------
__device__ __forceinline__ v16h load_a_frag(const _Float16* __restrict__ A,
                                            int lda, int m0, int k0, int lane) {
  const int row = m0 + (lane & 15);
  const int hs  = lane >> 4;
  const _Float16* p = A + (size_t)row * lda + k0;
  v8h lo = *(const v8h*)(p + hs * 8);
  v8h hi = *(const v8h*)(p + 16 + hs * 8);
  v16h r;
#pragma unroll
  for (int i = 0; i < 8; ++i) { r[i] = lo[i]; r[8 + i] = hi[i]; }
  return r;
}

__device__ __forceinline__ v16h load_bt_frag(const _Float16* __restrict__ Bt,
                                             int ldb, int n0, int k0, int lane) {
  const int col = n0 + (lane & 15);
  const int hs  = lane >> 4;
  const _Float16* p = Bt + (size_t)col * ldb + k0;
  v8h lo = *(const v8h*)(p + hs * 8);
  v8h hi = *(const v8h*)(p + 16 + hs * 8);
  v16h r;
#pragma unroll
  for (int i = 0; i < 8; ++i) { r[i] = lo[i]; r[8 + i] = hi[i]; }
  return r;
}

__device__ __forceinline__ v8f wmma_f16(v16h a, v16h b, v8f c) {
  return __builtin_amdgcn_wmma_f32_16x16x32_f16(false, a, false, b, (short)0, c,
                                                false, false);
}

// 32x64 wave tile, register double-buffered GEMM body. Prefetch k-step i+1's
// fragments, then issue k-step i's 8 WMMAs -> compiler can keep prefetch loads
// in flight across the WMMA batch instead of s_wait_loadcnt 0.
#define GEMM_BODY_32x64(APTR, LDA, BPTR, LDB, KDIM)                          \
  v16h a0 = load_a_frag(APTR, LDA, m0, 0, lane);                             \
  v16h a1 = load_a_frag(APTR, LDA, m0 + 16, 0, lane);                        \
  v16h b0 = load_bt_frag(BPTR, LDB, n0, 0, lane);                            \
  v16h b1 = load_bt_frag(BPTR, LDB, n0 + 16, 0, lane);                       \
  v16h b2 = load_bt_frag(BPTR, LDB, n0 + 32, 0, lane);                       \
  v16h b3 = load_bt_frag(BPTR, LDB, n0 + 48, 0, lane);                       \
  for (int k0 = 32; k0 < (KDIM); k0 += 32) {                                 \
    v16h na0 = load_a_frag(APTR, LDA, m0, k0, lane);                         \
    v16h na1 = load_a_frag(APTR, LDA, m0 + 16, k0, lane);                    \
    v16h nb0 = load_bt_frag(BPTR, LDB, n0, k0, lane);                        \
    v16h nb1 = load_bt_frag(BPTR, LDB, n0 + 16, k0, lane);                   \
    v16h nb2 = load_bt_frag(BPTR, LDB, n0 + 32, k0, lane);                   \
    v16h nb3 = load_bt_frag(BPTR, LDB, n0 + 48, k0, lane);                   \
    acc[0][0] = wmma_f16(a0, b0, acc[0][0]);                                 \
    acc[0][1] = wmma_f16(a0, b1, acc[0][1]);                                 \
    acc[0][2] = wmma_f16(a0, b2, acc[0][2]);                                 \
    acc[0][3] = wmma_f16(a0, b3, acc[0][3]);                                 \
    acc[1][0] = wmma_f16(a1, b0, acc[1][0]);                                 \
    acc[1][1] = wmma_f16(a1, b1, acc[1][1]);                                 \
    acc[1][2] = wmma_f16(a1, b2, acc[1][2]);                                 \
    acc[1][3] = wmma_f16(a1, b3, acc[1][3]);                                 \
    a0 = na0; a1 = na1; b0 = nb0; b1 = nb1; b2 = nb2; b3 = nb3;              \
  }                                                                          \
  acc[0][0] = wmma_f16(a0, b0, acc[0][0]);                                   \
  acc[0][1] = wmma_f16(a0, b1, acc[0][1]);                                   \
  acc[0][2] = wmma_f16(a0, b2, acc[0][2]);                                   \
  acc[0][3] = wmma_f16(a0, b3, acc[0][3]);                                   \
  acc[1][0] = wmma_f16(a1, b0, acc[1][0]);                                   \
  acc[1][1] = wmma_f16(a1, b1, acc[1][1]);                                   \
  acc[1][2] = wmma_f16(a1, b2, acc[1][2]);                                   \
  acc[1][3] = wmma_f16(a1, b3, acc[1][3]);

// ---------------------------------------------------------------------------
// 1) Weight prep: f32 [K,N] -> f16 [N,K] (transposed) for contiguous B frags.
// ---------------------------------------------------------------------------
__global__ __launch_bounds__(256) void k_prep_w(const float* __restrict__ Wqkv,
                                                const float* __restrict__ Wfc,
                                                _Float16* __restrict__ WtQkv,
                                                _Float16* __restrict__ WtFc) {
  int i = blockIdx.x * 256 + threadIdx.x;              // 0 .. 1048575
  const int NQ = QKV_N * IN_DIM;                       // 786432
  if (i < NQ) {
    int n = i / IN_DIM, k = i - n * IN_DIM;
    WtQkv[i] = (_Float16)Wqkv[(size_t)k * QKV_N + n];
  } else {
    int j = i - NQ;                                    // 0 .. 262143
    int n = j / IN_DIM, k = j - n * IN_DIM;
    WtFc[j] = (_Float16)Wfc[(size_t)k * IN_DIM + n];
  }
}

// ---------------------------------------------------------------------------
// 2) LayerNorm over last dim, output f16 row-major [NROWS, 512].
// ---------------------------------------------------------------------------
__global__ __launch_bounds__(256) void k_layernorm(const float* __restrict__ x,
                                                   const float* __restrict__ g,
                                                   const float* __restrict__ be,
                                                   _Float16* __restrict__ xn) {
  const int row = blockIdx.x;
  const int tid = threadIdx.x;
  const float* p = x + (size_t)row * IN_DIM;
  float v0 = p[tid], v1 = p[tid + 256];
  __shared__ float rs[256], rs2[256];
  rs[tid] = v0 + v1;
  rs2[tid] = v0 * v0 + v1 * v1;
  __syncthreads();
#pragma unroll
  for (int off = 128; off > 0; off >>= 1) {
    if (tid < off) { rs[tid] += rs[tid + off]; rs2[tid] += rs2[tid + off]; }
    __syncthreads();
  }
  const float mean = rs[0] * (1.0f / IN_DIM);
  const float var  = rs2[0] * (1.0f / IN_DIM) - mean * mean;
  const float rstd = rsqrtf(var + LN_EPS);
  _Float16* q = xn + (size_t)row * IN_DIM;
  q[tid]       = (_Float16)((v0 - mean) * rstd * g[tid] + be[tid]);
  q[tid + 256] = (_Float16)((v1 - mean) * rstd * g[tid + 256] + be[tid + 256]);
}

// ---------------------------------------------------------------------------
// 3) QKV GEMM: [16384,512] x [512,1536]. Wave tile 32x64 (8 wmma / k-step).
//    Q,K stored row-major f16; V stored transposed Vt[b][d][s] f16.
// ---------------------------------------------------------------------------
__global__ __launch_bounds__(256) void k_gemm_qkv(const _Float16* __restrict__ xn,
                                                  const _Float16* __restrict__ Wt,
                                                  _Float16* __restrict__ Q,
                                                  _Float16* __restrict__ Kk,
                                                  _Float16* __restrict__ Vt) {
  const int lane = threadIdx.x & 31;
  const int wave = threadIdx.x >> 5;
  const int m0 = blockIdx.x * 64 + (wave >> 2) * 32;
  const int n0 = blockIdx.y * 256 + (wave & 3) * 64;
  v8f acc[2][4] = {};
  GEMM_BODY_32x64(xn, IN_DIM, Wt, IN_DIM, IN_DIM)
  const int col = lane & 15, hs = lane >> 4;
#pragma unroll
  for (int i = 0; i < 2; ++i) {
#pragma unroll
    for (int t = 0; t < 4; ++t) {
      const int n = n0 + 16 * t + col;
#pragma unroll
      for (int r = 0; r < 8; ++r) {
        const int m = m0 + 16 * i + 8 * hs + r;
        const _Float16 v = (_Float16)acc[i][t][r];
        if (n < IN_DIM) {
          Q[(size_t)m * IN_DIM + n] = v;
        } else if (n < 2 * IN_DIM) {
          Kk[(size_t)m * IN_DIM + (n - IN_DIM)] = v;
        } else {
          const int b = m >> 11, s = m & (SEQ - 1);
          Vt[((size_t)b * IN_DIM + (n - 2 * IN_DIM)) * SEQ + s] = v;
        }
      }
    }
  }
}

// ---------------------------------------------------------------------------
// 4) Scores: S[b] = scale * Q[b] @ K[b]^T.  K rows ARE the transposed-B
//    operand ([n][k] contiguous). f16 store, [b][sq][sk].
// ---------------------------------------------------------------------------
__global__ __launch_bounds__(256) void k_scores(const _Float16* __restrict__ Q,
                                                const _Float16* __restrict__ Kk,
                                                _Float16* __restrict__ S) {
  const int b = blockIdx.z;
  const _Float16* Qb = Q + (size_t)b * SEQ * IN_DIM;
  const _Float16* Kb = Kk + (size_t)b * SEQ * IN_DIM;
  _Float16* Sb = S + (size_t)b * SEQ * SEQ;
  const int lane = threadIdx.x & 31;
  const int wave = threadIdx.x >> 5;
  const int m0 = blockIdx.x * 64 + (wave >> 2) * 32;
  const int n0 = blockIdx.y * 256 + (wave & 3) * 64;
  v8f acc[2][4] = {};
  GEMM_BODY_32x64(Qb, IN_DIM, Kb, IN_DIM, IN_DIM)
  const float scale = 0.044194173824159216f;  // 512^-0.5
  const int col = lane & 15, hs = lane >> 4;
#pragma unroll
  for (int i = 0; i < 2; ++i) {
#pragma unroll
    for (int t = 0; t < 4; ++t) {
      const int n = n0 + 16 * t + col;
#pragma unroll
      for (int r = 0; r < 8; ++r) {
        const int m = m0 + 16 * i + 8 * hs + r;
        Sb[(size_t)m * SEQ + n] = (_Float16)(acc[i][t][r] * scale);
      }
    }
  }
}

// ---------------------------------------------------------------------------
// 5) Row softmax stats: max and 1/sum(exp) per (b,sq).
// ---------------------------------------------------------------------------
__global__ __launch_bounds__(256) void k_stats(const _Float16* __restrict__ S,
                                               float* __restrict__ mrow,
                                               float* __restrict__ rlrow) {
  const int row = blockIdx.x;
  const int tid = threadIdx.x;
  const _Float16* p = S + (size_t)row * SEQ;
  __shared__ float red[256];
  float mx = -3.4e38f;
#pragma unroll
  for (int i = 0; i < 8; ++i) mx = fmaxf(mx, (float)p[tid + 256 * i]);
  red[tid] = mx;
  __syncthreads();
#pragma unroll
  for (int off = 128; off > 0; off >>= 1) {
    if (tid < off) red[tid] = fmaxf(red[tid], red[tid + off]);
    __syncthreads();
  }
  const float m = red[0];
  __syncthreads();
  float sum = 0.0f;
#pragma unroll
  for (int i = 0; i < 8; ++i) sum += __expf((float)p[tid + 256 * i] - m);
  red[tid] = sum;
  __syncthreads();
#pragma unroll
  for (int off = 128; off > 0; off >>= 1) {
    if (tid < off) red[tid] += red[tid + off];
    __syncthreads();
  }
  if (tid == 0) { mrow[row] = m; rlrow[row] = 1.0f / red[0]; }
}

// ---------------------------------------------------------------------------
// 6) O = softmax(S) @ V.  A fragment built on the fly: exp(s-m)/l, f16.
//    16x64 wave tile; raw score + B loads double-buffered; exps (VALU/TRANS)
//    can co-execute with the WMMA batch.
// ---------------------------------------------------------------------------
__global__ __launch_bounds__(256) void k_pv(const _Float16* __restrict__ S,
                                            const float* __restrict__ mrow,
                                            const float* __restrict__ rlrow,
                                            const _Float16* __restrict__ Vt,
                                            _Float16* __restrict__ O) {
  const int b = blockIdx.z;
  const int lane = threadIdx.x & 31;
  const int wave = threadIdx.x >> 5;
  const int m0 = blockIdx.x * 32 + (wave >> 2) * 16;   // query within batch
  const int n0 = blockIdx.y * 256 + (wave & 3) * 64;   // feature
  const int hs = lane >> 4;
  const int grow = b * SEQ + m0 + (lane & 15);
  const float mq = mrow[grow];
  const float rlq = rlrow[grow];
  const _Float16* ps = S + (size_t)grow * SEQ;
  const _Float16* Vb = Vt + (size_t)b * IN_DIM * SEQ;
  v8f acc[4] = {};

  v8h lo = *(const v8h*)(ps + hs * 8);
  v8h hi = *(const v8h*)(ps + 16 + hs * 8);
  v16h b0 = load_bt_frag(Vb, SEQ, n0, 0, lane);
  v16h b1 = load_bt_frag(Vb, SEQ, n0 + 16, 0, lane);
  v16h b2 = load_bt_frag(Vb, SEQ, n0 + 32, 0, lane);
  v16h b3 = load_bt_frag(Vb, SEQ, n0 + 48, 0, lane);
  for (int k0 = 32; k0 < SEQ; k0 += 32) {
    v8h nlo = *(const v8h*)(ps + k0 + hs * 8);
    v8h nhi = *(const v8h*)(ps + k0 + 16 + hs * 8);
    v16h nb0 = load_bt_frag(Vb, SEQ, n0, k0, lane);
    v16h nb1 = load_bt_frag(Vb, SEQ, n0 + 16, k0, lane);
    v16h nb2 = load_bt_frag(Vb, SEQ, n0 + 32, k0, lane);
    v16h nb3 = load_bt_frag(Vb, SEQ, n0 + 48, k0, lane);
    v16h a;
#pragma unroll
    for (int i = 0; i < 8; ++i) {
      a[i]     = (_Float16)(__expf((float)lo[i] - mq) * rlq);
      a[8 + i] = (_Float16)(__expf((float)hi[i] - mq) * rlq);
    }
    acc[0] = wmma_f16(a, b0, acc[0]);
    acc[1] = wmma_f16(a, b1, acc[1]);
    acc[2] = wmma_f16(a, b2, acc[2]);
    acc[3] = wmma_f16(a, b3, acc[3]);
    lo = nlo; hi = nhi; b0 = nb0; b1 = nb1; b2 = nb2; b3 = nb3;
  }
  {
    v16h a;
#pragma unroll
    for (int i = 0; i < 8; ++i) {
      a[i]     = (_Float16)(__expf((float)lo[i] - mq) * rlq);
      a[8 + i] = (_Float16)(__expf((float)hi[i] - mq) * rlq);
    }
    acc[0] = wmma_f16(a, b0, acc[0]);
    acc[1] = wmma_f16(a, b1, acc[1]);
    acc[2] = wmma_f16(a, b2, acc[2]);
    acc[3] = wmma_f16(a, b3, acc[3]);
  }
  const int col = lane & 15;
#pragma unroll
  for (int t = 0; t < 4; ++t) {
    const int n = n0 + 16 * t + col;
#pragma unroll
    for (int r = 0; r < 8; ++r) {
      const int m = m0 + 8 * hs + r;
      O[((size_t)b * SEQ + m) * IN_DIM + n] = (_Float16)acc[t][r];
    }
  }
}

// ---------------------------------------------------------------------------
// 7) y = LeakyReLU(O @ Wfc + b), f32 output.
// ---------------------------------------------------------------------------
__global__ __launch_bounds__(256) void k_fc(const _Float16* __restrict__ O,
                                            const _Float16* __restrict__ Wt,
                                            const float* __restrict__ bias,
                                            float* __restrict__ out) {
  const int lane = threadIdx.x & 31;
  const int wave = threadIdx.x >> 5;
  const int m0 = blockIdx.x * 64 + (wave >> 2) * 32;
  const int n0 = blockIdx.y * 256 + (wave & 3) * 64;
  v8f acc[2][4] = {};
  GEMM_BODY_32x64(O, IN_DIM, Wt, IN_DIM, IN_DIM)
  const int col = lane & 15, hs = lane >> 4;
#pragma unroll
  for (int i = 0; i < 2; ++i) {
#pragma unroll
    for (int t = 0; t < 4; ++t) {
      const int n = n0 + 16 * t + col;
      const float bn = bias[n];
#pragma unroll
      for (int r = 0; r < 8; ++r) {
        const int m = m0 + 16 * i + 8 * hs + r;
        float y = acc[i][t][r] + bn;
        out[(size_t)m * IN_DIM + n] = (y >= 0.0f) ? y : LEAKY * y;
      }
    }
  }
}

// ---------------------------------------------------------------------------
extern "C" void kernel_launch(void* const* d_in, const int* in_sizes, int n_in,
                              void* d_out, int out_size, void* d_ws, size_t ws_size,
                              hipStream_t stream) {
  const float* x    = (const float*)d_in[0];
  const float* g    = (const float*)d_in[1];
  const float* be   = (const float*)d_in[2];
  const float* Wqkv = (const float*)d_in[3];
  const float* Wfc  = (const float*)d_in[4];
  const float* bfc  = (const float*)d_in[5];
  float* out = (float*)d_out;

  char* w = (char*)d_ws;
  _Float16* WtQkv = (_Float16*)w; w += (size_t)QKV_N * IN_DIM * 2;     // 1.5 MB
  _Float16* WtFc  = (_Float16*)w; w += (size_t)IN_DIM * IN_DIM * 2;    // 0.5 MB
  _Float16* xn    = (_Float16*)w; w += (size_t)NROWS * IN_DIM * 2;     // 16 MB
  _Float16* Q     = (_Float16*)w; w += (size_t)NROWS * IN_DIM * 2;     // 16 MB
  _Float16* Kk    = (_Float16*)w; w += (size_t)NROWS * IN_DIM * 2;     // 16 MB
  _Float16* Vt    = (_Float16*)w; w += (size_t)NROWS * IN_DIM * 2;     // 16 MB
  _Float16* Sc    = (_Float16*)w; w += (size_t)BATCH * SEQ * SEQ * 2;  // 64 MB
  float*    mrow  = (float*)w;    w += (size_t)NROWS * 4;
  float*    rlrow = (float*)w;    w += (size_t)NROWS * 4;
  _Float16* O     = (_Float16*)w; w += (size_t)NROWS * IN_DIM * 2;     // 16 MB

  k_prep_w   <<<dim3(4096), 256, 0, stream>>>(Wqkv, Wfc, WtQkv, WtFc);
  k_layernorm<<<dim3(NROWS), 256, 0, stream>>>(x, g, be, xn);
  k_gemm_qkv <<<dim3(NROWS / 64, QKV_N / 256), 256, 0, stream>>>(xn, WtQkv, Q, Kk, Vt);
  k_scores   <<<dim3(SEQ / 64, SEQ / 256, BATCH), 256, 0, stream>>>(Q, Kk, Sc);
  k_stats    <<<dim3(NROWS), 256, 0, stream>>>(Sc, mrow, rlrow);
  k_pv       <<<dim3(SEQ / 32, IN_DIM / 256, BATCH), 256, 0, stream>>>(Sc, mrow, rlrow, Vt, O);
  k_fc       <<<dim3(NROWS / 64, IN_DIM / 256), 256, 0, stream>>>(O, WtFc, bfc, out);
}
